// MultiQueryAttention_21122649162484
// MI455X (gfx1250) — compile-verified, matrix operand order checked by reading, not run on hardware
//
#include <hip/hip_runtime.h>
#include <math.h>

typedef _Float16 v16h __attribute__((ext_vector_type(16)));
typedef _Float16 v8h  __attribute__((ext_vector_type(8)));
typedef _Float16 v4h  __attribute__((ext_vector_type(4)));
typedef float    v8f  __attribute__((ext_vector_type(8)));

union HFrag { v16h v; v8h h[2]; };

// Load one WMMA A/B fragment half-pair for this lane.
// `p` must already point at: row_base + (lane>=16 ? 8 : 0).
__device__ __forceinline__ v16h load_frag(const _Float16* p) {
    HFrag f;
    f.h[0] = *reinterpret_cast<const v8h*>(p);
    f.h[1] = *reinterpret_cast<const v8h*>(p + 16);
    return f.v;
}

__device__ __forceinline__ v8f wmma_f16(v16h a, v16h b, v8f c) {
    return __builtin_amdgcn_wmma_f32_16x16x32_f16(false, a, false, b,
                                                  (short)0, c, false, false);
}

// gfx1250 async VMEM->LDS copy (16B per lane), tracked by ASYNCcnt.
__device__ __forceinline__ void async_copy16(unsigned int lds_byte_addr,
                                             const void* gptr) {
    unsigned long long ga = (unsigned long long)(uintptr_t)gptr;
    asm volatile("global_load_async_to_lds_b128 %0, %1, off"
                 :: "v"(lds_byte_addr), "v"(ga) : "memory");
}
__device__ __forceinline__ void wait_async0() {
    asm volatile("s_wait_asynccnt 0x0" ::: "memory");
}
__device__ __forceinline__ unsigned int lds_addr_of(const void* p) {
    return (unsigned int)(uintptr_t)p;   // LDS aperture: low 32 bits = LDS byte addr
}

// ---------------------------------------------------------------------------
// Prep kernels: one-time fp32 -> fp16 conversion / weight transpose.
// ---------------------------------------------------------------------------
__global__ __launch_bounds__(256) void cvt_f16_kernel(
    const float* __restrict__ in, _Float16* __restrict__ out, int n4)
{
    int i = blockIdx.x * 256 + threadIdx.x;
    if (i < n4) {
        float4 f = reinterpret_cast<const float4*>(in)[i];
        v4h o = { (_Float16)f.x, (_Float16)f.y, (_Float16)f.z, (_Float16)f.w };
        reinterpret_cast<v4h*>(out)[i] = o;
    }
}

// W (Kd x Nd, f32, row-major) -> WT (Nd x Kd, f16, row-major)
__global__ __launch_bounds__(256) void transpose_cvt_kernel(
    const float* __restrict__ W, _Float16* __restrict__ WT, int Kd, int Nd)
{
    __shared__ _Float16 tile[32][33];
    const int tx = threadIdx.x, ty = threadIdx.y;     // 32 x 8
    const int n0 = blockIdx.x * 32, k0 = blockIdx.y * 32;
#pragma unroll
    for (int i = 0; i < 4; ++i)
        tile[ty + i * 8][tx] = (_Float16)W[(size_t)(k0 + ty + i * 8) * Nd + n0 + tx];
    __syncthreads();
#pragma unroll
    for (int i = 0; i < 4; ++i)
        WT[(size_t)(n0 + ty + i * 8) * Kd + k0 + tx] = tile[tx][ty + i * 8];
}

// ---------------------------------------------------------------------------
// WMMA GEMM, f16 operands: C = alpha * (A @ BT^T)
//   A : M x K f16 row-major          BT : N x K f16 row-major (pre-transposed)
// Double-buffered LDS staging via global_load_async_to_lds_b128.
// modes: 0 = f16 row-major out   1 = f32 row-major out
//        2 = kv split: col<64 -> K (b,t,64); col>=64 -> V^T (b,64,T)
// ---------------------------------------------------------------------------
#define BM 128
#define BN 128
#define BK 32
#define LP 40   // LDS row pitch in halves (80B keeps 16B alignment of K-groups)

__global__ __launch_bounds__(256) void wmma_gemm_f16(
    const _Float16* __restrict__ A, const _Float16* __restrict__ BT,
    int M, int K, int N,
    float alpha, void* __restrict__ Cp, int mode, int Tdim)
{
    __shared__ alignas(16) _Float16 As[2][BM * LP];
    __shared__ alignas(16) _Float16 Bs[2][BN * LP];

    const int tid  = threadIdx.x;
    const int lane = tid & 31;
    const int wave = tid >> 5;
    const int wm = wave & 3;            // 4 waves along M
    const int wn = wave >> 2;           // 2 waves along N
    const int bm0 = blockIdx.y * BM;
    const int bn0 = blockIdx.x * BN;
    const int l15 = lane & 15;
    const int hi8 = (lane >> 4) << 3;

    // each thread copies two 16B chunks of A and two of B per tile
    const int srow = tid >> 2;              // 0..63 (plus +64 on second iter)
    const int skg  = (tid & 3) * 8;         // K-group within 32

    v8f acc[2][4];
#pragma unroll
    for (int mi = 0; mi < 2; ++mi)
#pragma unroll
        for (int ni = 0; ni < 4; ++ni)
            acc[mi][ni] = {};

    auto stage = [&](int buf, int k0) {
#pragma unroll
        for (int i = 0; i < 2; ++i) {
            const int row = srow + i * 64;
            async_copy16(lds_addr_of(&As[buf][row * LP + skg]),
                         &A[(size_t)(bm0 + row) * K + k0 + skg]);
        }
#pragma unroll
        for (int i = 0; i < 2; ++i) {
            const int row = srow + i * 64;
            async_copy16(lds_addr_of(&Bs[buf][row * LP + skg]),
                         &BT[(size_t)(bn0 + row) * K + k0 + skg]);
        }
    };

    stage(0, 0);
    wait_async0();
    __syncthreads();

    int buf = 0;
    for (int k0 = 0; k0 < K; k0 += BK) {
        if (k0 + BK < K) stage(buf ^ 1, k0 + BK);   // prefetch next tile

        v16h aF[2], bF[4];
#pragma unroll
        for (int mi = 0; mi < 2; ++mi)
            aF[mi] = load_frag(&As[buf][(wm * 32 + mi * 16 + l15) * LP + hi8]);
#pragma unroll
        for (int ni = 0; ni < 4; ++ni)
            bF[ni] = load_frag(&Bs[buf][(wn * 64 + ni * 16 + l15) * LP + hi8]);

#pragma unroll
        for (int mi = 0; mi < 2; ++mi)
#pragma unroll
            for (int ni = 0; ni < 4; ++ni)
                acc[mi][ni] = wmma_f16(aF[mi], bF[ni], acc[mi][ni]);

        wait_async0();       // next tile landed in LDS
        __syncthreads();     // all waves done reading `buf` / writing `buf^1`
        buf ^= 1;
    }

    // ---- epilogue ----
#pragma unroll
    for (int mi = 0; mi < 2; ++mi) {
#pragma unroll
        for (int ni = 0; ni < 4; ++ni) {
            const int rbase = bm0 + wm * 32 + mi * 16 + hi8;
            const int col   = bn0 + wn * 64 + ni * 16 + l15;
#pragma unroll
            for (int j = 0; j < 8; ++j) {
                const int row = rbase + j;
                const float val = acc[mi][ni][j] * alpha;
                if (mode == 0) {
                    ((_Float16*)Cp)[(size_t)row * N + col] = (_Float16)val;
                } else if (mode == 1) {
                    ((float*)Cp)[(size_t)row * N + col] = val;
                } else {
                    _Float16* kout = (_Float16*)Cp;
                    _Float16* vout = kout + (size_t)M * 64;
                    const int b = row / Tdim;
                    const int t = row - b * Tdim;
                    if (col < 64)
                        kout[(size_t)row * 64 + col] = (_Float16)val;
                    else
                        vout[(size_t)(b * 64 + (col - 64)) * Tdim + t] = (_Float16)val;
                }
            }
        }
    }
}

// ---------------------------------------------------------------------------
// Flash-style multi-query attention. One wave handles 16 query rows of one
// (b, h) over the full T, chunked 32 keys at a time. q pre-scaled by
// hd^-0.5 * log2(e) so softmax uses exp2 (v_exp_f32).
// ---------------------------------------------------------------------------
__global__ __launch_bounds__(128) void mqa_attn_kernel(
    const _Float16* __restrict__ q, const _Float16* __restrict__ kmat,
    const _Float16* __restrict__ vT, _Float16* __restrict__ y,
    int Nq, int Tlen, int D, int Hn)
{
    __shared__ alignas(16) _Float16 Ps[4 * 16 * LP];

    const int lane = threadIdx.x & 31;
    const int wv   = threadIdx.x >> 5;
    const int l15  = lane & 15;
    const int hi8  = (lane >> 4) << 3;

    const int qtiles = Nq >> 4;
    int gw = blockIdx.x * 4 + wv;
    const int qt = gw % qtiles; gw /= qtiles;
    const int h  = gw % Hn;
    const int b  = gw / Hn;

    const _Float16* qbase = q + ((size_t)(b * Nq + qt * 16 + l15)) * D + h * 64;
    v16h aQ[2];
    aQ[0] = load_frag(qbase + 0  + hi8);
    aQ[1] = load_frag(qbase + 32 + hi8);

    const _Float16* kbase = kmat + (size_t)b * Tlen * 64 + hi8;
    const _Float16* vbase = vT   + (size_t)(b * 64) * Tlen + hi8;

    float m[8], l[8];
    v8f accY[4];
#pragma unroll
    for (int j = 0; j < 8; ++j) { m[j] = -1e30f; l[j] = 0.0f; }
#pragma unroll
    for (int ni = 0; ni < 4; ++ni) accY[ni] = {};

    _Float16* ps = &Ps[wv * 16 * LP];

    for (int tc = 0; tc < Tlen; tc += 32) {
        v8f s0 = {}, s1 = {};
        {
            const _Float16* kp0 = kbase + (size_t)(tc + l15) * 64;
            const _Float16* kp1 = kbase + (size_t)(tc + 16 + l15) * 64;
            v16h b00 = load_frag(kp0);
            v16h b01 = load_frag(kp0 + 32);
            v16h b10 = load_frag(kp1);
            v16h b11 = load_frag(kp1 + 32);
            s0 = wmma_f16(aQ[0], b00, s0);
            s0 = wmma_f16(aQ[1], b01, s0);
            s1 = wmma_f16(aQ[0], b10, s1);
            s1 = wmma_f16(aQ[1], b11, s1);
        }

#pragma unroll
        for (int j = 0; j < 8; ++j) {
            float mt = fmaxf(s0[j], s1[j]);
#pragma unroll
            for (int off = 8; off >= 1; off >>= 1)
                mt = fmaxf(mt, __shfl_xor(mt, off, 16));
            const float mn = fmaxf(m[j], mt);
            const float sc = exp2f(m[j] - mn);
            const float p0 = exp2f(s0[j] - mn);
            const float p1 = exp2f(s1[j] - mn);
            float rs = p0 + p1;
#pragma unroll
            for (int off = 8; off >= 1; off >>= 1)
                rs += __shfl_xor(rs, off, 16);
            l[j] = l[j] * sc + rs;
            m[j] = mn;
            s0[j] = p0; s1[j] = p1;
            accY[0][j] *= sc; accY[1][j] *= sc;
            accY[2][j] *= sc; accY[3][j] *= sc;
        }

        // C-layout -> LDS -> A-layout for P (same-wave LDS RAW is in-order)
#pragma unroll
        for (int j = 0; j < 8; ++j) {
            const int row = j + hi8;
            ps[row * LP + l15]      = (_Float16)s0[j];
            ps[row * LP + 16 + l15] = (_Float16)s1[j];
        }
        v16h aP = load_frag(&ps[l15 * LP + hi8]);

#pragma unroll
        for (int ni = 0; ni < 4; ++ni) {
            const _Float16* vp = vbase + (size_t)(ni * 16 + l15) * Tlen + tc;
            v16h bV = load_frag(vp);
            accY[ni] = wmma_f16(aP, bV, accY[ni]);
        }
    }

    float inv[8];
#pragma unroll
    for (int j = 0; j < 8; ++j) inv[j] = 1.0f / l[j];

    const size_t obase = ((size_t)(b * Nq + qt * 16)) * D + h * 64;
#pragma unroll
    for (int ni = 0; ni < 4; ++ni)
#pragma unroll
        for (int j = 0; j < 8; ++j) {
            const int row = j + hi8;
            y[obase + (size_t)row * D + ni * 16 + l15] =
                (_Float16)(accY[ni][j] * inv[j]);
        }
}

// ---------------------------------------------------------------------------
extern "C" void kernel_launch(void* const* d_in, const int* in_sizes, int n_in,
                              void* d_out, int out_size, void* d_ws, size_t ws_size,
                              hipStream_t stream) {
    const float* x   = (const float*)d_in[0];
    const float* ctx = (const float*)d_in[1];
    const float* Wq  = (const float*)d_in[2];
    const float* Wkv = (const float*)d_in[3];
    const float* Wo  = (const float*)d_in[4];
    float* out = (float*)d_out;

    const int Bc = 2, Nc = 2048, Tc = 2048, Dc = 1024, Hc = 16;
    const int MQ  = Bc * Nc;   // 4096
    const int MKV = Bc * Tc;   // 4096

    // workspace (f16): x16 | ctx16 | WqT | WkvT | WoT | q | k | vT | y
    _Float16* x16   = (_Float16*)d_ws;
    _Float16* ctx16 = x16   + (size_t)MQ * Dc;
    _Float16* WqT   = ctx16 + (size_t)MKV * Dc;
    _Float16* WkvT  = WqT   + (size_t)Dc * Dc;
    _Float16* WoT   = WkvT  + (size_t)128 * Dc;
    _Float16* qbuf  = WoT   + (size_t)Dc * Dc;
    _Float16* kb    = qbuf  + (size_t)MQ * Dc;
    _Float16* vT    = kb    + (size_t)MKV * 64;
    _Float16* y     = vT    + (size_t)MKV * 64;

    const float alphaQ = 0.125f * 1.44269504088896340736f;  // hd^-0.5 * log2(e)

    // ---- prep: fp32 -> fp16 (+ weight transposes) ----
    cvt_f16_kernel<<<dim3((MQ * Dc / 4 + 255) / 256), dim3(256), 0, stream>>>(
        x, x16, MQ * Dc / 4);
    cvt_f16_kernel<<<dim3((MKV * Dc / 4 + 255) / 256), dim3(256), 0, stream>>>(
        ctx, ctx16, MKV * Dc / 4);
    transpose_cvt_kernel<<<dim3(Dc / 32, Dc / 32), dim3(32, 8), 0, stream>>>(
        Wq, WqT, Dc, Dc);
    transpose_cvt_kernel<<<dim3(128 / 32, Dc / 32), dim3(32, 8), 0, stream>>>(
        Wkv, WkvT, Dc, 128);
    transpose_cvt_kernel<<<dim3(Dc / 32, Dc / 32), dim3(32, 8), 0, stream>>>(
        Wo, WoT, Dc, Dc);

    dim3 blk(256);
    // q = (x @ Wq) * alphaQ -> f16
    wmma_gemm_f16<<<dim3(Dc / 128, MQ / 128), blk, 0, stream>>>(
        x16, WqT, MQ, Dc, Dc, alphaQ, qbuf, 0, 0);
    // kv = ctx @ Wkv -> k row-major + v transposed
    wmma_gemm_f16<<<dim3(1, MKV / 128), blk, 0, stream>>>(
        ctx16, WkvT, MKV, Dc, 128, 1.0f, kb, 2, Tc);
    // attention -> y (f16)
    const int totalWaves = Bc * Hc * (Nc / 16);   // 4096
    mqa_attn_kernel<<<dim3(totalWaves / 4), dim3(128), 0, stream>>>(
        qbuf, kb, vT, y, Nc, Tc, Dc, Hc);
    // out = y @ Wo -> f32
    wmma_gemm_f16<<<dim3(Dc / 128, MQ / 128), blk, 0, stream>>>(
        y, WoT, MQ, Dc, Dc, 1.0f, out, 1, 0);
}
